// SpliceGraph_27393301414017
// MI455X (gfx1250) — matrix operand order repaired and checked by reading, not compiled, not run on hardware
//
#include <hip/hip_runtime.h>
#include <hip/hip_bf16.h>
#include <math.h>

#define N_NODES 50000
#define N_EDGES 800000
#define D_IN    64
#define D_HID   256
#define BN_EPS  1e-5f

typedef __attribute__((ext_vector_type(16))) __bf16 v16bf;
typedef __attribute__((ext_vector_type(8)))  float  v8f;

union ABf { v16bf v; unsigned u[8]; };

// ---------------- utility kernels ----------------
__global__ void cvt_f32_bf16(const float* __restrict__ in, __bf16* __restrict__ out, int n) {
  int i = blockIdx.x * blockDim.x + threadIdx.x;
  if (i < n) out[i] = (__bf16)in[i];
}

// WT[n*K + k] = W[k*Ncol + n]  (bf16 transpose so WMMA B-fragments load coalesced)
__global__ void cvt_transpose_bf16(const float* __restrict__ W, __bf16* __restrict__ WT,
                                   int K, int Ncol) {
  int i = blockIdx.x * blockDim.x + threadIdx.x;
  if (i < K * Ncol) {
    int k = i / Ncol, n = i % Ncol;
    WT[n * K + k] = (__bf16)W[i];
  }
}

__global__ void deg_init_kernel(float* deg, int n) {
  int i = blockIdx.x * blockDim.x + threadIdx.x;
  if (i < n) deg[i] = 1.0f;            // self loop
}

__global__ void deg_count_kernel(const int* __restrict__ dst, float* deg, int e) {
  int i = blockIdx.x * blockDim.x + threadIdx.x;
  if (i < e) atomicAdd(&deg[dst[i]], 1.0f);
}

__global__ void dinv_kernel(float* deg, int n) {
  int i = blockIdx.x * blockDim.x + threadIdx.x;
  if (i < n) deg[i] = rsqrtf(fmaxf(deg[i], 1.0f));   // in-place -> dinv
}

__global__ void zero_kernel(float* p, int n) {
  int i = blockIdx.x * blockDim.x + threadIdx.x;
  if (i < n) p[i] = 0.0f;
}

// ---------------- fused input GEMMs: H = xs@W_gcn, XL = xs@W_lin + b_lin ----------------
__global__ __launch_bounds__(256) void gemm_in_kernel(
    const __bf16* __restrict__ A,    // [N, 64] bf16
    const __bf16* __restrict__ BgT,  // [256, 64] = W_gcn^T bf16
    const __bf16* __restrict__ BlT,  // [256, 64] = W_lin^T bf16
    const float*  __restrict__ b_lin,
    float* __restrict__ H, float* __restrict__ XL)
{
  int wave = (blockIdx.x * blockDim.x + threadIdx.x) >> 5;
  int lane = threadIdx.x & 31;
  int half = lane >> 4, m = lane & 15;
  const int colTiles = D_HID / 16;
  int rt = wave / colTiles, ct = wave % colTiles;
  if (rt >= N_NODES / 16) return;
  int row = rt * 16 + m;
  int col = ct * 16 + m;
  const unsigned* a32  = (const unsigned*)(A   + (size_t)row * D_IN);
  const unsigned* bg32 = (const unsigned*)(BgT + (size_t)col * D_IN);
  const unsigned* bl32 = (const unsigned*)(BlT + (size_t)col * D_IN);
  v8f cg = {}; v8f cl = {};
#pragma unroll
  for (int k0 = 0; k0 < D_IN; k0 += 32) {
    ABf a, bg, bl;
#pragma unroll
    for (int r = 0; r < 8; ++r) {
      // A 16-bit fragment: lane=M row; half 0 -> K {0..7,16..23}, half 1 -> K {8..15,24..31}
      a.u[r]  = a32[(k0 >> 1) + 4 * half + r + (r < 4 ? 0 : 4)];
      // B 16-bit fragment (from transposed W): lane=N col; half 0 -> K {0..7,16..23}, half 1 -> K {8..15,24..31}
      bg.u[r] = bg32[(k0 >> 1) + 4 * half + r + (r < 4 ? 0 : 4)];
      bl.u[r] = bl32[(k0 >> 1) + 4 * half + r + (r < 4 ? 0 : 4)];
    }
    cg = __builtin_amdgcn_wmma_f32_16x16x32_bf16(false, a.v, false, bg.v, (short)0, cg, false, false);
    cl = __builtin_amdgcn_wmma_f32_16x16x32_bf16(false, a.v, false, bl.v, (short)0, cl, false, false);
  }
  float blb = b_lin[col];
#pragma unroll
  for (int d = 0; d < 8; ++d) {
    int orow = rt * 16 + d + 8 * half;           // C layout: VGPR d -> M=d (+8 for upper half)
    size_t idx = (size_t)orow * D_HID + col;
    H[idx]  = cg[d];
    XL[idx] = cl[d] + blb;
  }
}

// ---------------- self loop: Z = H * dinv^2 ----------------
__global__ void zinit_kernel(const float* __restrict__ H, const float* __restrict__ dinv,
                             float* __restrict__ Z) {
  size_t i = (size_t)blockIdx.x * blockDim.x + threadIdx.x;
  if (i < (size_t)N_NODES * D_HID) {
    float di = dinv[i >> 8];
    Z[i] = H[i] * di * di;
  }
}

// ---------------- edge aggregation: Z[dst] += H[src] * dinv[src]*dinv[dst] ----------------
__global__ __launch_bounds__(256) void edge_kernel(const int* __restrict__ src,
                                                   const int* __restrict__ dst,
                                                   const float* __restrict__ H,
                                                   const float* __restrict__ dinv,
                                                   float* Z) {
  int nw   = (gridDim.x * blockDim.x) >> 5;
  int wave = (blockIdx.x * blockDim.x + threadIdx.x) >> 5;
  int lane = threadIdx.x & 31;
  for (int e = wave; e < N_EDGES; e += nw) {
    int s = src[e], d = dst[e];
    int e2 = e + nw;
    if (e2 < N_EDGES) __builtin_prefetch(H + (size_t)src[e2] * D_HID, 0, 1);
    float nrm = dinv[s] * dinv[d];
    const float4* hs = (const float4*)(H + (size_t)s * D_HID);
    float* zd = Z + (size_t)d * D_HID;
    float4 v0 = hs[lane];
    float4 v1 = hs[lane + 32];
    int c0 = lane * 4, c1 = (lane + 32) * 4;
    atomicAdd(&zd[c0 + 0], v0.x * nrm); atomicAdd(&zd[c0 + 1], v0.y * nrm);
    atomicAdd(&zd[c0 + 2], v0.z * nrm); atomicAdd(&zd[c0 + 3], v0.w * nrm);
    atomicAdd(&zd[c1 + 0], v1.x * nrm); atomicAdd(&zd[c1 + 1], v1.y * nrm);
    atomicAdd(&zd[c1 + 2], v1.z * nrm); atomicAdd(&zd[c1 + 3], v1.w * nrm);
  }
}

// ---------------- Z = tanh(Z + b_gcn); also emit bf16 copy for gate GEMM ----------------
__global__ void ztanh_kernel(float* Z, const float* __restrict__ b_gcn,
                             __bf16* __restrict__ Zb) {
  size_t i = (size_t)blockIdx.x * blockDim.x + threadIdx.x;
  if (i < (size_t)N_NODES * D_HID) {
    float z = tanhf(Z[i] + b_gcn[i & 255]);
    Z[i]  = z;
    Zb[i] = (__bf16)z;
  }
}

// ---------------- gate GEMM + sigmoid + gated residual + ReLU ----------------
__global__ __launch_bounds__(256) void gate_kernel(
    const __bf16* __restrict__ Zb,   // [N,256] bf16
    const __bf16* __restrict__ WgT,  // [256,256] = W_gate^T bf16
    const float*  __restrict__ b_gate,
    const float*  __restrict__ Z,    // fp32 tanh(z)
    const float*  __restrict__ XL,
    float* __restrict__ OUT)
{
  int wave = (blockIdx.x * blockDim.x + threadIdx.x) >> 5;
  int lane = threadIdx.x & 31;
  int half = lane >> 4, m = lane & 15;
  int rt = wave / 16, ct = wave % 16;
  if (rt >= N_NODES / 16) return;
  int row = rt * 16 + m;
  int col = ct * 16 + m;
  const unsigned* a32 = (const unsigned*)(Zb  + (size_t)row * D_HID);
  const unsigned* b32 = (const unsigned*)(WgT + (size_t)col * D_HID);
  v8f acc = {};
#pragma unroll
  for (int k0 = 0; k0 < D_HID; k0 += 32) {
    ABf a, b;
#pragma unroll
    for (int r = 0; r < 8; ++r) {
      a.u[r] = a32[(k0 >> 1) + 4 * half + r + (r < 4 ? 0 : 4)];
      b.u[r] = b32[(k0 >> 1) + 4 * half + r + (r < 4 ? 0 : 4)];
    }
    acc = __builtin_amdgcn_wmma_f32_16x16x32_bf16(false, a.v, false, b.v, (short)0, acc, false, false);
  }
  float bg = b_gate[col];
#pragma unroll
  for (int d = 0; d < 8; ++d) {
    size_t idx = (size_t)(rt * 16 + d + 8 * half) * D_HID + col;
    float g = 1.0f / (1.0f + __expf(-(acc[d] + bg)));
    float v = (1.0f - g) * XL[idx] + g * Z[idx];
    OUT[idx] = fmaxf(v, 0.0f);
  }
}

// ---------------- per-column sum / sumsq (BN stats) ----------------
__global__ __launch_bounds__(256) void colstats_kernel(const float* __restrict__ OUT,
                                                       float* colsum, float* colsq) {
  int c = threadIdx.x;                     // one column per thread, 256 threads
  float s = 0.0f, sq = 0.0f;
  for (int r = blockIdx.x; r < N_NODES; r += gridDim.x) {
    float v = OUT[(size_t)r * D_HID + c];
    s += v; sq += v * v;
  }
  atomicAdd(&colsum[c], s);
  atomicAdd(&colsq[c], sq);
}

// ---------------- BN finalize (in-place on OUT) ----------------
__global__ void bn_kernel(float* OUT, const float* __restrict__ colsum,
                          const float* __restrict__ colsq,
                          const float* __restrict__ gamma,
                          const float* __restrict__ beta) {
  size_t i = (size_t)blockIdx.x * blockDim.x + threadIdx.x;
  if (i < (size_t)N_NODES * D_HID) {
    int c = i & 255;
    const float invN = 1.0f / (float)N_NODES;
    float mu  = colsum[c] * invN;
    float var = colsq[c] * invN - mu * mu;
    OUT[i] = (OUT[i] - mu) * rsqrtf(var + BN_EPS) * gamma[c] + beta[c];
  }
}

// ---------------- host launcher ----------------
extern "C" void kernel_launch(void* const* d_in, const int* in_sizes, int n_in,
                              void* d_out, int out_size, void* d_ws, size_t ws_size,
                              hipStream_t stream) {
  const float* xs     = (const float*)d_in[0];
  const int*   ei     = (const int*)d_in[1];
  const int*   src    = ei;
  const int*   dst    = ei + N_EDGES;
  const float* W_gcn  = (const float*)d_in[2];
  const float* b_gcn  = (const float*)d_in[3];
  const float* W_lin  = (const float*)d_in[4];
  const float* b_lin  = (const float*)d_in[5];
  const float* W_gate = (const float*)d_in[6];
  const float* b_gate = (const float*)d_in[7];
  const float* gamma  = (const float*)d_in[8];
  const float* beta   = (const float*)d_in[9];
  float* OUT = (float*)d_out;

  char* p = (char*)d_ws;
  auto alloc = [&](size_t bytes) -> char* {
    char* r = p; p += (bytes + 255) & ~(size_t)255; return r;
  };
  __bf16* xsb    = (__bf16*)alloc((size_t)N_NODES * D_IN * 2);
  __bf16* WgT    = (__bf16*)alloc((size_t)D_IN  * D_HID * 2);
  __bf16* WlT    = (__bf16*)alloc((size_t)D_IN  * D_HID * 2);
  __bf16* WgateT = (__bf16*)alloc((size_t)D_HID * D_HID * 2);
  float*  H      = (float*) alloc((size_t)N_NODES * D_HID * 4);
  float*  XL     = (float*) alloc((size_t)N_NODES * D_HID * 4);
  float*  Z      = (float*) alloc((size_t)N_NODES * D_HID * 4);
  __bf16* Zb     = (__bf16*)alloc((size_t)N_NODES * D_HID * 2);
  float*  deg    = (float*) alloc((size_t)N_NODES * 4);          // becomes dinv
  float*  colsum = (float*) alloc((size_t)D_HID * 4);
  float*  colsq  = (float*) alloc((size_t)D_HID * 4);
  (void)ws_size; (void)n_in; (void)in_sizes; (void)out_size;

  const int T = 256;
  // precision / layout prep
  cvt_f32_bf16<<<(N_NODES * D_IN + T - 1) / T, T, 0, stream>>>(xs, xsb, N_NODES * D_IN);
  cvt_transpose_bf16<<<(D_IN * D_HID + T - 1) / T, T, 0, stream>>>(W_gcn, WgT, D_IN, D_HID);
  cvt_transpose_bf16<<<(D_IN * D_HID + T - 1) / T, T, 0, stream>>>(W_lin, WlT, D_IN, D_HID);
  cvt_transpose_bf16<<<(D_HID * D_HID + T - 1) / T, T, 0, stream>>>(W_gate, WgateT, D_HID, D_HID);

  // degree -> dinv
  deg_init_kernel<<<(N_NODES + T - 1) / T, T, 0, stream>>>(deg, N_NODES);
  deg_count_kernel<<<(N_EDGES + T - 1) / T, T, 0, stream>>>(dst, deg, N_EDGES);
  dinv_kernel<<<(N_NODES + T - 1) / T, T, 0, stream>>>(deg, N_NODES);

  // fused input projections (WMMA bf16)
  const int tiles = (N_NODES / 16) * (D_HID / 16);     // 50000 waves
  gemm_in_kernel<<<tiles / 8, T, 0, stream>>>(xsb, WgT, WlT, b_lin, H, XL);

  // aggregation
  zinit_kernel<<<(int)(((size_t)N_NODES * D_HID + T - 1) / T), T, 0, stream>>>(H, deg, Z);
  edge_kernel<<<2048, T, 0, stream>>>(src, dst, H, deg, Z);
  ztanh_kernel<<<(int)(((size_t)N_NODES * D_HID + T - 1) / T), T, 0, stream>>>(Z, b_gcn, Zb);

  // gate GEMM + combine (WMMA bf16)
  gate_kernel<<<tiles / 8, T, 0, stream>>>(Zb, WgateT, b_gate, Z, XL, OUT);

  // batch norm
  zero_kernel<<<1, T, 0, stream>>>(colsum, D_HID);
  zero_kernel<<<1, T, 0, stream>>>(colsq, D_HID);
  colstats_kernel<<<256, T, 0, stream>>>(OUT, colsum, colsq);
  bn_kernel<<<(int)(((size_t)N_NODES * D_HID + T - 1) / T), T, 0, stream>>>(OUT, colsum, colsq, gamma, beta);
}
